// GConvGRUCell_43258910605776
// MI455X (gfx1250) — matrix operand order repaired and checked by reading, not compiled, not run on hardware
//
#include <hip/hip_runtime.h>
#include <hip/hip_bf16.h>

// Problem constants (match reference)
#define B_   4
#define N_   20000
#define E_   320000
#define CIN_ 64
#define H_   128
#define K_   192      // C = CIN + H
#define OZR_ 256      // 2H
#define OH_  128
#define KT_  6        // K / 32 wmma steps
#define M_   (B_ * N_)   // 80000 rows, divisible by 16

typedef __attribute__((ext_vector_type(16))) __bf16 v16bf;
typedef __attribute__((ext_vector_type(8)))  float  v8f;

union FragBF { uint4 u[2]; v16bf v; };

// ---------------------------------------------------------------- degree / norm
__global__ void k_deg(const int* __restrict__ dst, float* __restrict__ deg) {
    int e = blockIdx.x * 256 + threadIdx.x;
    if (e < E_) atomicAdd(&deg[dst[e]], 1.0f);
}

__global__ void k_dinv(float* __restrict__ deg_dinv) {
    int n = blockIdx.x * 256 + threadIdx.x;
    if (n < N_) deg_dinv[n] = rsqrtf(deg_dinv[n] + 1.0f);  // +1 self loop
}

__global__ void k_norm(const int* __restrict__ src, const int* __restrict__ dst,
                       const float* __restrict__ dinv, float* __restrict__ norm) {
    int e = blockIdx.x * 256 + threadIdx.x;
    if (e < E_) norm[e] = dinv[src[e]] * dinv[dst[e]];
}

// ---------------------------------------------------------------- feature build
__global__ void k_build_xh(const float* __restrict__ x, const float* __restrict__ h,
                           __hip_bfloat16* __restrict__ fa) {
    int idx = blockIdx.x * 256 + threadIdx.x;           // over M_*K_
    if (idx >= M_ * K_) return;
    int c  = idx % K_;
    int bn = idx / K_;
    float v = (c < CIN_) ? x[(size_t)bn * CIN_ + c]
                         : h[(size_t)bn * H_ + (c - CIN_)];
    fa[idx] = __float2bfloat16(v);
}

// ---------------------------------------------------------------- W pre-pack into
// wave32 B-fragment layout: lane = (Khalf<<4)|col, 16 contiguous bf16 = K run of 16.
__global__ void k_pack_w(const float* __restrict__ W,  // [K_, O] row-major
                         __hip_bfloat16* __restrict__ Wp, int O, int total) {
    int idx = blockIdx.x * 256 + threadIdx.x;           // ((nt*KT_+kt)*32+lane)*16+j
    if (idx >= total) return;
    int j    = idx & 15;
    int lane = (idx >> 4) & 31;
    int kt   = (idx >> 9) % KT_;
    int nt   = idx / (KT_ * 512);
    int n = nt * 16 + (lane & 15);
    int k = kt * 32 + (lane >> 4) * 16 + j;
    Wp[idx] = __float2bfloat16(W[(size_t)k * O + n]);
}

// ---------------------------------------------------------------- WMMA GEMM
// One wave per 16x16 tile of Y = A(bf16 [M_,K_]) * W(prepacked), f32 accumulate.
__global__ __launch_bounds__(128)
void k_gemm_bf16(const __hip_bfloat16* __restrict__ A,
                 const __hip_bfloat16* __restrict__ Wp,
                 float* __restrict__ Y,
                 int O, int nTilesN, int nTiles) {
    const int lane = threadIdx.x & 31;
    const int wave = threadIdx.x >> 5;
    const int tile = blockIdx.x * 4 + wave;
    if (tile >= nTiles) return;
    const int mt   = tile / nTilesN;
    const int nt   = tile - mt * nTilesN;
    const int half = lane >> 4;                      // K-half selector
    const int row  = mt * 16 + (lane & 15);          // A row for this lane

    const __hip_bfloat16* arow  = A + (size_t)row * K_;
    const __hip_bfloat16* bbase = Wp + ((size_t)(nt * KT_) * 32 + lane) * 16;

    v8f acc;
    #pragma unroll
    for (int i = 0; i < 8; ++i) acc[i] = 0.0f;

    #pragma unroll
    for (int kt = 0; kt < KT_; ++kt) {
        FragBF a, b;
        // A 16x32 layout (matches packed B): lane half h holds K = [16h,16h+16)
        a.u[0] = *(const uint4*)(arow + kt * 32 + half * 16);
        a.u[1] = *(const uint4*)(arow + kt * 32 + half * 16 + 8);
        const uint4* bp = (const uint4*)(bbase + (size_t)kt * 512);
        b.u[0] = bp[0];
        b.u[1] = bp[1];
        acc = __builtin_amdgcn_wmma_f32_16x16x32_bf16(
                  false, a.v, false, b.v, (short)0, acc, false, false);
    }

    // C/D layout: VGPR v -> row (mt*16 + v + 8*half), col = nt*16 + (lane&15)
    float* yrow = Y + (size_t)(mt * 16 + 8 * half) * O + nt * 16 + (lane & 15);
    #pragma unroll
    for (int v = 0; v < 8; ++v) yrow[(size_t)v * O] = acc[v];
}

// ---------------------------------------------------------------- aggregation
// agg = bias + lin * dinv^2   (fuses self-loop message + bias)
__global__ void k_agg_init(const float* __restrict__ lin, const float* __restrict__ bias,
                           const float* __restrict__ dinv, float* __restrict__ agg, int O) {
    int O4  = O >> 2;
    int idx = blockIdx.x * 256 + threadIdx.x;        // over M_*O4
    if (idx >= M_ * O4) return;
    int c4 = idx % O4;
    int bn = idx / O4;
    int n  = bn % N_;
    float di = dinv[n];
    float w  = di * di;
    const float4 lv = *(const float4*)(lin  + (size_t)bn * O + c4 * 4);
    const float4 bv = *(const float4*)(bias + (size_t)c4 * 4);
    float4 r;
    r.x = bv.x + lv.x * w;  r.y = bv.y + lv.y * w;
    r.z = bv.z + lv.z * w;  r.w = bv.w + lv.w * w;
    *(float4*)(agg + (size_t)bn * O + c4 * 4) = r;
}

// agg[b,dst] += lin[b,src] * norm   (L2-resident gather + f32 atomics)
__global__ void k_scatter(const float* __restrict__ lin, float* __restrict__ agg,
                          const int* __restrict__ src, const int* __restrict__ dst,
                          const float* __restrict__ norm, int O) {
    int O4  = O >> 2;
    int idx = blockIdx.x * 256 + threadIdx.x;        // over E_*O4 (< 2^31)
    if (idx >= E_ * O4) return;
    int b  = blockIdx.y;
    int e  = idx / O4;
    int c4 = idx - e * O4;
    int s = src[e], d = dst[e];
    float w = norm[e];
    const float4 v = *(const float4*)(lin + ((size_t)b * N_ + s) * O + c4 * 4);
    float* o = agg + ((size_t)b * N_ + d) * O + c4 * 4;
    atomicAdd(o + 0, v.x * w);
    atomicAdd(o + 1, v.y * w);
    atomicAdd(o + 2, v.z * w);
    atomicAdd(o + 3, v.w * w);
}

// ---------------------------------------------------------------- gates
// z = sigmoid(zr[:,:H]) stored in place; feature h-part <- bf16(r*h)
__global__ void k_gate(float* __restrict__ aggzr, const float* __restrict__ h,
                       __hip_bfloat16* __restrict__ fa) {
    int idx = blockIdx.x * 256 + threadIdx.x;        // over M_*H_
    if (idx >= M_ * H_) return;
    int j  = idx % H_;
    int bn = idx / H_;
    float* zr = aggzr + (size_t)bn * OZR_;
    float z = 1.0f / (1.0f + expf(-zr[j]));
    float r = 1.0f / (1.0f + expf(-zr[H_ + j]));
    zr[j] = z;
    float rh = r * h[(size_t)bn * H_ + j];
    fa[(size_t)bn * K_ + CIN_ + j] = __float2bfloat16(rh);
}

__global__ void k_final(const float* __restrict__ aggzr, const float* __restrict__ aggh,
                        const float* __restrict__ h, float* __restrict__ out) {
    int idx = blockIdx.x * 256 + threadIdx.x;        // over M_*H_
    if (idx >= M_ * H_) return;
    int j  = idx % H_;
    int bn = idx / H_;
    float z  = aggzr[(size_t)bn * OZR_ + j];
    float ht = tanhf(aggh[(size_t)bn * H_ + j]);
    float hv = h[idx];
    out[idx] = (1.0f - z) * hv + z * ht;
}

// ---------------------------------------------------------------- launcher
extern "C" void kernel_launch(void* const* d_in, const int* in_sizes, int n_in,
                              void* d_out, int out_size, void* d_ws, size_t ws_size,
                              hipStream_t stream) {
    const float* x   = (const float*)d_in[0];
    const float* h   = (const float*)d_in[1];
    const float* Wzr = (const float*)d_in[2];
    const float* bzr = (const float*)d_in[3];
    const float* Wh  = (const float*)d_in[4];
    const float* bh  = (const float*)d_in[5];
    const int*   ei  = (const int*)d_in[6];
    const int* src = ei;        // edge_index[0]
    const int* dst = ei + E_;   // edge_index[1]
    float* out = (float*)d_out;

    // workspace carve-out (~196 MB), 256B aligned regions
    char* wsp = (char*)d_ws;
    auto carve = [&](size_t bytes) {
        char* p = wsp;
        wsp += (bytes + 255) & ~(size_t)255;
        return p;
    };
    float*          dinv  = (float*)carve((size_t)N_ * 4);            // deg -> dinv in place
    float*          norm  = (float*)carve((size_t)E_ * 4);
    __hip_bfloat16* featA = (__hip_bfloat16*)carve((size_t)M_ * K_ * 2);
    __hip_bfloat16* WpZr  = (__hip_bfloat16*)carve((size_t)16 * KT_ * 512 * 2);
    __hip_bfloat16* WpH   = (__hip_bfloat16*)carve((size_t)8  * KT_ * 512 * 2);
    float*          lin   = (float*)carve((size_t)M_ * OZR_ * 4);     // also lin_h + agg_h
    float*          aggzr = (float*)carve((size_t)M_ * OZR_ * 4);
    float*          linh  = lin;                                      // [M_,128] in lower half
    float*          aggh  = lin + (size_t)M_ * OH_;                   // upper half

    auto cdiv = [](long long a, long long b) { return (unsigned)((a + b - 1) / b); };

    // 1) degrees & norms
    hipMemsetAsync(dinv, 0, (size_t)N_ * 4, stream);
    k_deg <<<cdiv(E_, 256), 256, 0, stream>>>(dst, dinv);
    k_dinv<<<cdiv(N_, 256), 256, 0, stream>>>(dinv);
    k_norm<<<cdiv(E_, 256), 256, 0, stream>>>(src, dst, dinv, norm);

    // 2) features + weight packing
    k_build_xh<<<cdiv((long long)M_ * K_, 256), 256, 0, stream>>>(x, h, featA);
    k_pack_w<<<cdiv(16 * KT_ * 512, 256), 256, 0, stream>>>(Wzr, WpZr, OZR_, 16 * KT_ * 512);
    k_pack_w<<<cdiv(8  * KT_ * 512, 256), 256, 0, stream>>>(Wh,  WpH,  OH_,  8  * KT_ * 512);

    // 3) GCN #1: lin = [x,h] @ W_zr ; agg = b + self-loop + edge scatter
    {
        int nTiles = (M_ / 16) * (OZR_ / 16);   // 80000
        k_gemm_bf16<<<cdiv(nTiles, 4), 128, 0, stream>>>(featA, WpZr, lin, OZR_, OZR_ / 16, nTiles);
    }
    k_agg_init<<<cdiv((long long)M_ * (OZR_ / 4), 256), 256, 0, stream>>>(lin, bzr, dinv, aggzr, OZR_);
    k_scatter<<<dim3(cdiv((long long)E_ * (OZR_ / 4), 256), B_), 256, 0, stream>>>(
        lin, aggzr, src, dst, norm, OZR_);

    // 4) gates: z in place, featA h-part <- r*h
    k_gate<<<cdiv((long long)M_ * H_, 256), 256, 0, stream>>>(aggzr, h, featA);

    // 5) GCN #2: lin_h = [x, r*h] @ W_h ; aggregate
    {
        int nTiles = (M_ / 16) * (OH_ / 16);    // 40000
        k_gemm_bf16<<<cdiv(nTiles, 4), 128, 0, stream>>>(featA, WpH, linh, OH_, OH_ / 16, nTiles);
    }
    k_agg_init<<<cdiv((long long)M_ * (OH_ / 4), 256), 256, 0, stream>>>(linh, bh, dinv, aggh, OH_);
    k_scatter<<<dim3(cdiv((long long)E_ * (OH_ / 4), 256), B_), 256, 0, stream>>>(
        linh, aggh, src, dst, norm, OH_);

    // 6) output combine
    k_final<<<cdiv((long long)M_ * H_, 256), 256, 0, stream>>>(aggzr, aggh, h, out);
}
